// Graph_Transformer_11484742549826
// MI455X (gfx1250) — matrix operand order repaired and checked by reading, not compiled, not run on hardware
//
#include <hip/hip_runtime.h>
#include <math.h>

// ---------------------------------------------------------------------------
// DETR-style graph transformer forward for MI455X (gfx1250), bf16 WMMA compute
// Packed bf16-pair LDS tiles + 128-bit ds loads; clamped (branchless) loads;
// global_prefetch for streamed GEMM tiles; uniform fast-path epilogues.
// ---------------------------------------------------------------------------

#define D_MODEL 256
#define FF      2048
#define NHEAD   8
#define DH      32
#define BATCH   4
#define NE      6
#define ND      4
#define LRELU   0.01f
#define ROWSTR  (BATCH * D_MODEL)   // 1024 floats: power-of-two row stride

typedef __attribute__((ext_vector_type(16))) __bf16       v16bf;
typedef __attribute__((ext_vector_type(2)))  __bf16       v2bf;
typedef __attribute__((ext_vector_type(8)))  float        v8f;
typedef __attribute__((ext_vector_type(2)))  float        v2f;
typedef __attribute__((ext_vector_type(8)))  unsigned int v8u;

#define WMMA_BF16(a, b, c) \
    __builtin_amdgcn_wmma_f32_16x16x32_bf16(false, (a), false, (b), (short)0, (c), false, false)

static __device__ __forceinline__ unsigned int pkbf16(float lo, float hi) {
    v2f v = {lo, hi};
    return __builtin_bit_cast(unsigned int, __builtin_convertvector(v, v2bf));
}
static __device__ __forceinline__ float4 ld4(const float* p) {
    return *(const float4*)p;
}
static __device__ __forceinline__ v16bf mk_frag(uint4 a, uint4 b) {
    v8u u = {a.x, a.y, a.z, a.w, b.x, b.y, b.z, b.w};
    return __builtin_bit_cast(v16bf, u);
}
static __device__ __forceinline__ v16bf mk_frag8(const unsigned int* w) {
    v8u u = {w[0], w[1], w[2], w[3], w[4], w[5], w[6], w[7]};
    return __builtin_bit_cast(v16bf, u);
}
static __device__ __forceinline__ float lrelu(float v, int act) {
    return act ? ((v > 0.f) ? v : LRELU * v) : v;
}

// ---------------------------------------------------------------------------
// GEMM: C[M,N] = act(A[M,K] @ W[K,N] + bias[N]); fp32 in/out, bf16 WMMA.
// Block = 64x64 tile, 256 threads = 8 wave32s; wave = 16 rows x 32 cols.
// LDS packed bf16 k-pairs; clamped streaming loads; next-tile prefetch.
// ---------------------------------------------------------------------------
__global__ __launch_bounds__(256) void gemm_bias_act_kernel(
    const float* __restrict__ A, const float* __restrict__ W,
    const float* __restrict__ bias, float* __restrict__ C,
    int M, int K, int N, int act)
{
    __shared__ alignas(16) unsigned int As2[64][20];
    __shared__ alignas(16) unsigned int Bs2[64][20];

    const int row0 = blockIdx.x * 64;
    const int col0 = blockIdx.y * 64;
    const int t    = threadIdx.x;
    const int lane = t & 31;
    const int wave = t >> 5;
    const int wr   = (wave & 3) * 16;
    const int wc   = (wave >> 2) * 32;
    const int n16  = lane & 15;
    const int g    = lane >> 4;

    // loader geometry (fixed per thread)
    const int ar  = t >> 4;          // A tile row base: rows ar + 16*i
    const int akp = t & 15;          // A k-pair index
    const int bkp = t >> 6;          // B k-pair base: pairs bkp + 4*i
    const int bc  = t & 63;          // B col

    // clamped streaming global pointers (row clamp -> branchless loads;
    // garbage rows only affect discarded output rows)
    const float* ap0 = A + (size_t)min(row0 + ar,      M - 1) * K + 2 * akp;
    const float* ap1 = A + (size_t)min(row0 + ar + 16, M - 1) * K + 2 * akp;
    const float* ap2 = A + (size_t)min(row0 + ar + 32, M - 1) * K + 2 * akp;
    const float* ap3 = A + (size_t)min(row0 + ar + 48, M - 1) * K + 2 * akp;
    const float* bp  = W + (size_t)(2 * bkp) * N + col0 + bc;

    // loop-invariant LDS fragment pointers
    const uint4* pa  = (const uint4*)&As2[wr + n16][0];
    const uint4* pb0 = (const uint4*)&Bs2[wc + n16][0];
    const uint4* pb1 = (const uint4*)&Bs2[wc + 16 + n16][0];

    v8f acc0 = {0.f,0.f,0.f,0.f,0.f,0.f,0.f,0.f};
    v8f acc1 = {0.f,0.f,0.f,0.f,0.f,0.f,0.f,0.f};

    for (int k0 = 0; k0 < K; k0 += 32) {
        __syncthreads();
        {
            float2 v0 = *(const float2*)ap0;
            float2 v1 = *(const float2*)ap1;
            float2 v2 = *(const float2*)ap2;
            float2 v3 = *(const float2*)ap3;
            As2[ar     ][akp] = pkbf16(v0.x, v0.y);
            As2[ar + 16][akp] = pkbf16(v1.x, v1.y);
            As2[ar + 32][akp] = pkbf16(v2.x, v2.y);
            As2[ar + 48][akp] = pkbf16(v3.x, v3.y);
            // prefetch next k-tile (speculative; safe past the end)
            __builtin_prefetch(ap0 + 32, 0, 0);
            __builtin_prefetch(ap1 + 32, 0, 0);
            __builtin_prefetch(ap2 + 32, 0, 0);
            __builtin_prefetch(ap3 + 32, 0, 0);
            ap0 += 32; ap1 += 32; ap2 += 32; ap3 += 32;
        }
        {
#pragma unroll
            for (int i = 0; i < 4; ++i) {
                const float* p = bp + (size_t)(8 * i) * N;
                Bs2[bc][bkp + 4 * i] = pkbf16(p[0], p[N]);
            }
            __builtin_prefetch(bp + (size_t)32 * N, 0, 0);
            bp += (size_t)32 * N;
        }
        __syncthreads();

        v16bf af  = mk_frag(pa[g], pa[2 + g]);           // words g*4.., 8+g*4..
        v16bf bf0 = mk_frag(pb0[g], pb0[2 + g]);         // words g*4.., 8+g*4..
        v16bf bf1 = mk_frag(pb1[g], pb1[2 + g]);
        acc0 = WMMA_BF16(af, bf0, acc0);
        acc1 = WMMA_BF16(af, bf1, acc1);
    }

    // epilogue: loop-invariant bias, streaming row pointer, uniform fast path
    const int gc0 = col0 + wc + n16;
    const int gc1 = gc0 + 16;
    const float b0 = bias[gc0];
    const float b1 = bias[gc1];
    float* crow = C + (size_t)(row0 + wr + 8 * g) * N;
    if (row0 + 64 <= M) {
#pragma unroll
        for (int r8 = 0; r8 < 8; ++r8) {
            crow[gc0] = lrelu(acc0[r8] + b0, act);
            crow[gc1] = lrelu(acc1[r8] + b1, act);
            crow += N;
        }
    } else {
#pragma unroll
        for (int r8 = 0; r8 < 8; ++r8) {
            int gr = row0 + wr + 8 * g + r8;
            if (gr < M) {
                crow[gc0] = lrelu(acc0[r8] + b0, act);
                crow[gc1] = lrelu(acc1[r8] + b1, act);
            }
            crow += N;
        }
    }
}

// ---------------------------------------------------------------------------
// Flash attention: O = softmax(Q K^T / sqrt(dh)) V, one (b,h) + 16-query tile
// per wave32 block.  Buffers [N,BATCH,D] flattened: row = n*BATCH + b, head h
// at cols [h*32, h*32+32).  Row stride is 4 KB -> pure shift addressing.
// Loads are clamped (branchless); OOB keys are killed by the -1e9 score mask
// (expf underflow -> exactly-zero probability).
// ---------------------------------------------------------------------------
__global__ __launch_bounds__(32) void flash_attn_kernel(
    const float* __restrict__ Q, const float* __restrict__ K,
    const float* __restrict__ V, float* __restrict__ O,
    int Nq, int Nk)
{
    __shared__ alignas(16) unsigned int VtP[32][20]; // [dh col][key pair]
    __shared__ alignas(16) float        PtF[16][36]; // [q row][key in tile]

    const int lane = threadIdx.x;
    const int q0   = blockIdx.x * 16;
    const int bh   = blockIdx.y;
    const int b    = bh / NHEAD;
    const int h    = bh % NHEAD;
    const int n16  = lane & 15;
    const int g    = lane >> 4;
    const int kp_  = lane & 15;     // V-staging: key pair
    const int half = lane >> 4;     // V-staging: dh half
    const float scale = 0.17677669529663687f;  // 1/sqrt(32)

    const float* Qbase = Q + b * D_MODEL + h * DH;
    const float* Kbase = K + b * D_MODEL + h * DH;
    const float* Vbase = V + b * D_MODEL + h * DH;

    // ---- Q fragment: one contiguous 16-float run per lane (clamped row) ----
    v16bf aq;
    {
        int qrow = min(q0 + n16, Nq - 1);
        const float* qp = Qbase + (size_t)qrow * ROWSTR;
        float f[16];
        *(float4*)(f + 0)  = ld4(qp + g * 16);
        *(float4*)(f + 4)  = ld4(qp + g * 16 + 4);
        *(float4*)(f + 8)  = ld4(qp + g * 16 + 8);
        *(float4*)(f + 12) = ld4(qp + g * 16 + 12);
        unsigned int w[8];
#pragma unroll
        for (int r8 = 0; r8 < 8; ++r8) w[r8] = pkbf16(f[2 * r8], f[2 * r8 + 1]);
        aq = mk_frag8(w);
    }

    // loop-invariant LDS fragment pointers
    const uint4*  pv0 = (const uint4*)&VtP[n16][0];
    const uint4*  pv1 = (const uint4*)&VtP[16 + n16][0];
    const float*  pp0 = &PtF[n16][0];

    v8f o0 = {0.f,0.f,0.f,0.f,0.f,0.f,0.f,0.f};
    v8f o1 = {0.f,0.f,0.f,0.f,0.f,0.f,0.f,0.f};
    float mrow[8], lrow[8];
#pragma unroll
    for (int r8 = 0; r8 < 8; ++r8) { mrow[r8] = -3.0e38f; lrow[r8] = 0.f; }

    for (int kt = 0; kt < Nk; kt += 32) {
        __syncthreads();   // previous-iteration LDS reads done

        // ---- stage V tile transposed-packed: VtP[col][kp] (clamped keys) ----
        {
            const float* vA = Vbase + (size_t)min(kt + 2 * kp_,     Nk - 1) * ROWSTR + half * 16;
            const float* vB = Vbase + (size_t)min(kt + 2 * kp_ + 1, Nk - 1) * ROWSTR + half * 16;
            float fa[16], fb[16];
            *(float4*)(fa + 0) = ld4(vA + 0);  *(float4*)(fa + 4)  = ld4(vA + 4);
            *(float4*)(fa + 8) = ld4(vA + 8);  *(float4*)(fa + 12) = ld4(vA + 12);
            *(float4*)(fb + 0) = ld4(vB + 0);  *(float4*)(fb + 4)  = ld4(vB + 4);
            *(float4*)(fb + 8) = ld4(vB + 8);  *(float4*)(fb + 12) = ld4(vB + 12);
#pragma unroll
            for (int j = 0; j < 16; ++j)
                VtP[half * 16 + j][kp_] = pkbf16(fa[j], fb[j]);
        }

        // ---- K fragments straight from global (clamped keys) ----
        v16bf bk0, bk1;
        {
            const float* kp = Kbase + (size_t)min(kt + n16, Nk - 1) * ROWSTR + g * 16;
            float f[16];
            *(float4*)(f + 0) = ld4(kp + 0);  *(float4*)(f + 4)  = ld4(kp + 4);
            *(float4*)(f + 8) = ld4(kp + 8);  *(float4*)(f + 12) = ld4(kp + 12);
            unsigned int w[8];
#pragma unroll
            for (int r8 = 0; r8 < 8; ++r8) w[r8] = pkbf16(f[2 * r8], f[2 * r8 + 1]);
            bk0 = mk_frag8(w);
        }
        {
            const float* kp = Kbase + (size_t)min(kt + 16 + n16, Nk - 1) * ROWSTR + g * 16;
            float f[16];
            *(float4*)(f + 0) = ld4(kp + 0);  *(float4*)(f + 4)  = ld4(kp + 4);
            *(float4*)(f + 8) = ld4(kp + 8);  *(float4*)(f + 12) = ld4(kp + 12);
            unsigned int w[8];
#pragma unroll
            for (int r8 = 0; r8 < 8; ++r8) w[r8] = pkbf16(f[2 * r8], f[2 * r8 + 1]);
            bk1 = mk_frag8(w);
        }

        v8f z = {0.f,0.f,0.f,0.f,0.f,0.f,0.f,0.f};
        v8f s0 = WMMA_BF16(aq, bk0, z);
        v8f s1 = WMMA_BF16(aq, bk1, z);

        bool ok0 = (kt + n16) < Nk;
        bool ok1 = (kt + 16 + n16) < Nk;
#pragma unroll
        for (int r8 = 0; r8 < 8; ++r8) {
            float a = ok0 ? s0[r8] * scale : -1e9f;
            float c = ok1 ? s1[r8] * scale : -1e9f;
            float mx = fmaxf(a, c);
            for (int mm = 1; mm < 16; mm <<= 1)
                mx = fmaxf(mx, __shfl_xor(mx, mm, 32));
            float mnew = fmaxf(mrow[r8], mx);
            float corr = __expf(mrow[r8] - mnew);
            float p0 = __expf(a - mnew);
            float p1 = __expf(c - mnew);
            float rs = p0 + p1;
            for (int mm = 1; mm < 16; mm <<= 1)
                rs += __shfl_xor(rs, mm, 32);
            lrow[r8] = lrow[r8] * corr + rs;
            mrow[r8] = mnew;
            o0[r8] *= corr;
            o1[r8] *= corr;
            PtF[r8 + 8 * g][n16]      = p0;   // C layout -> LDS (f32)
            PtF[r8 + 8 * g][16 + n16] = p1;
        }
        __syncthreads();

        // ---- P fragment (A layout) via contiguous float4 runs ----
        v16bf ap;
        {
            float f[16];
            *(float4*)(f + 0)  = *(const float4*)(pp0 + g * 8);
            *(float4*)(f + 4)  = *(const float4*)(pp0 + g * 8 + 4);
            *(float4*)(f + 8)  = *(const float4*)(pp0 + 16 + g * 8);
            *(float4*)(f + 12) = *(const float4*)(pp0 + 16 + g * 8 + 4);
            unsigned int w[8];
#pragma unroll
            for (int r8 = 0; r8 < 8; ++r8) w[r8] = pkbf16(f[2 * r8], f[2 * r8 + 1]);
            ap = mk_frag8(w);
        }
        // ---- V fragments: two ds_load_b128 each ----
        v16bf bv0 = mk_frag(pv0[g], pv0[2 + g]);
        v16bf bv1 = mk_frag(pv1[g], pv1[2 + g]);

        o0 = WMMA_BF16(ap, bv0, o0);
        o1 = WMMA_BF16(ap, bv1, o1);
    }

    // output: streaming row pointer, uniform fast path
    float* op = O + ((size_t)(q0 + 8 * g) * BATCH + b) * D_MODEL + h * DH;
    if (q0 + 16 <= Nq) {
#pragma unroll
        for (int r8 = 0; r8 < 8; ++r8) {
            float inv = 1.f / lrow[r8];
            op[n16]      = o0[r8] * inv;
            op[16 + n16] = o1[r8] * inv;
            op += ROWSTR;
        }
    } else {
#pragma unroll
        for (int r8 = 0; r8 < 8; ++r8) {
            if (q0 + 8 * g + r8 < Nq) {
                float inv = 1.f / lrow[r8];
                op[n16]      = o0[r8] * inv;
                op[16 + n16] = o1[r8] * inv;
            }
            op += ROWSTR;
        }
    }
}

// ---------------------------------------------------------------------------
// Fused residual + LayerNorm over D=256 (one row/block, 256 thr, wave shfl).
// ---------------------------------------------------------------------------
__global__ __launch_bounds__(256) void add_ln_kernel(
    const float* __restrict__ a, const float* __restrict__ b, int has_b,
    const float* __restrict__ gam, const float* __restrict__ bet,
    float* __restrict__ o)
{
    __shared__ float ws1[8], ws2[8];
    int r = blockIdx.x, t = threadIdx.x, ln = t & 31, wv = t >> 5;
    float x = a[(size_t)r * D_MODEL + t];
    if (has_b) x += b[(size_t)r * D_MODEL + t];
    float s = x, s2 = x * x;
    for (int mm = 1; mm < 32; mm <<= 1) {
        s  += __shfl_xor(s,  mm, 32);
        s2 += __shfl_xor(s2, mm, 32);
    }
    if (ln == 0) { ws1[wv] = s; ws2[wv] = s2; }
    __syncthreads();
    float tot = 0.f, tot2 = 0.f;
#pragma unroll
    for (int i = 0; i < 8; ++i) { tot += ws1[i]; tot2 += ws2[i]; }
    float mu  = tot * (1.f / D_MODEL);
    float var = tot2 * (1.f / D_MODEL) - mu * mu;
    o[(size_t)r * D_MODEL + t] = (x - mu) * rsqrtf(var + 1e-5f) * gam[t] + bet[t];
}

// ---------------------------------------------------------------------------
__global__ void add2_kernel(const float* __restrict__ a,
                            const float* __restrict__ b,
                            float* __restrict__ o, int n)
{
    int i = blockIdx.x * blockDim.x + threadIdx.x;
    if (i < n) o[i] = a[i] + b[i];
}

// ---------------------------------------------------------------------------
// Head-averaged attention probs for last decoder self-attention:
// edge[b,n,m] = mean_h softmax_m(q_bnh . k_bmh / sqrt(dh)).  (100x100, tiny)
// ---------------------------------------------------------------------------
__global__ __launch_bounds__(128) void edge_kernel(
    const float* __restrict__ Q, const float* __restrict__ K,
    float* __restrict__ edge, int Nq)
{
    __shared__ float qv[D_MODEL];
    __shared__ float wred[4];
    int n = blockIdx.x, b = blockIdx.y, t = threadIdx.x;
    int ln = t & 31, wv = t >> 5;
    qv[t]       = Q[((size_t)n * BATCH + b) * D_MODEL + t];
    qv[t + 128] = Q[((size_t)n * BATCH + b) * D_MODEL + t + 128];
    __syncthreads();

    const float scale = 0.17677669529663687f;
    float acc = 0.f;
    for (int h = 0; h < NHEAD; ++h) {
        float s = -3.0e38f;
        if (t < Nq) {
            const float* kr = K + ((size_t)t * BATCH + b) * D_MODEL + h * DH;
            const float* qr = qv + h * DH;
            float dp = 0.f;
#pragma unroll
            for (int d = 0; d < DH; ++d) dp += qr[d] * kr[d];
            s = dp * scale;
        }
        float mx = s;
        for (int mm = 1; mm < 32; mm <<= 1) mx = fmaxf(mx, __shfl_xor(mx, mm, 32));
        if (ln == 0) wred[wv] = mx;
        __syncthreads();
        mx = fmaxf(fmaxf(wred[0], wred[1]), fmaxf(wred[2], wred[3]));
        __syncthreads();
        float e = (t < Nq) ? __expf(s - mx) : 0.f;
        float su = e;
        for (int mm = 1; mm < 32; mm <<= 1) su += __shfl_xor(su, mm, 32);
        if (ln == 0) wred[wv] = su;
        __syncthreads();
        float tot = wred[0] + wred[1] + wred[2] + wred[3];
        __syncthreads();
        if (t < Nq) acc += e / tot;
    }
    if (t < Nq) edge[((size_t)b * Nq + n) * Nq + t] = acc * (1.f / NHEAD);
}

// ---------------------------------------------------------------------------
// Orchestration
// ---------------------------------------------------------------------------
extern "C" void kernel_launch(void* const* d_in, const int* in_sizes, int n_in,
                              void* d_out, int out_size, void* d_ws, size_t ws_size,
                              hipStream_t stream)
{
    (void)in_sizes; (void)n_in; (void)out_size; (void)ws_size;
    const int T = 1024, Nq = 100;
    const int NTB = T * BATCH;    // 4096 encoder rows
    const int NQB = Nq * BATCH;   // 400 decoder rows

    const float* src       = (const float*)d_in[0];
    const float* pos       = (const float*)d_in[1];
    const float* query_pos = (const float*)d_in[2];
    const float* tgt       = (const float*)d_in[3];
    const int ENC0 = 4;                 // 16 params per encoder layer
    const int DEC0 = ENC0 + NE * 16;    // 26 params per decoder layer
    const float* dn_s = (const float*)d_in[DEC0 + ND * 26];
    const float* dn_b = (const float*)d_in[DEC0 + ND * 26 + 1];

    // workspace layout (fp32)
    float* w = (float*)d_ws;
    size_t off = 0;
    auto alloc = [&](size_t nelem) { float* p = w + off; off += nelem; return p; };
    const size_t BIG   = (size_t)NTB * D_MODEL;
    const size_t SMALL = (size_t)NQB * D_MODEL;
    float* qin  = alloc(BIG);
    float* Qb   = alloc(BIG);
    float* Kb   = alloc(BIG);
    float* Vb   = alloc(BIG);
    float* att  = alloc(BIG);
    float* proj = alloc(BIG);
    float* x1   = alloc(BIG);
    float* memA = alloc(BIG);
    float* memB = alloc(BIG);
    float* mpos = alloc(BIG);
    float* hid  = alloc((size_t)NTB * FF);
    float* tA   = alloc(SMALL);
    float* tB   = alloc(SMALL);
    float* qind = alloc(SMALL);
    float* Qd   = alloc(SMALL);
    float* Kd   = alloc(SMALL);
    float* Vd   = alloc(SMALL);
    float* attd = alloc(SMALL);
    float* prjd = alloc(SMALL);
    float* x1d  = alloc(SMALL);
    float* x2d  = alloc(SMALL);
    float* hidd = alloc((size_t)NQB * FF);

    auto GEMM = [&](const float* A, const float* Wt, const float* bias, float* C,
                    int M, int K, int N, int act) {
        dim3 grid((M + 63) / 64, N / 64);
        gemm_bias_act_kernel<<<grid, 256, 0, stream>>>(A, Wt, bias, C, M, K, N, act);
    };
    auto FLASH = [&](const float* Qp, const float* Kp, const float* Vp, float* Op,
                     int nq, int nk) {
        dim3 grid((nq + 15) / 16, BATCH * NHEAD);
        flash_attn_kernel<<<grid, 32, 0, stream>>>(Qp, Kp, Vp, Op, nq, nk);
    };
    auto ADD = [&](const float* a, const float* b, float* o2, int n) {
        add2_kernel<<<(n + 255) / 256, 256, 0, stream>>>(a, b, o2, n);
    };
    auto ADDLN = [&](const float* a, const float* b, int hasb,
                     const float* gm, const float* bt, float* o2, int R) {
        add_ln_kernel<<<R, 256, 0, stream>>>(a, b, hasb, gm, bt, o2);
    };

    // ---------------- encoder ----------------
    const float* cur = src;
    for (int i = 0; i < NE; ++i) {
        const float* const* P = (const float* const*)(d_in + ENC0 + i * 16);
        // 0:Wq 1:bq 2:Wk 3:bk 4:Wv 5:bv 6:Wo 7:bo 8:ln1s 9:ln1b 10:ln2s 11:ln2b
        // 12:ff1W 13:ff1b 14:ff2W 15:ff2b
        float* outb = (i & 1) ? memB : memA;
        ADD(cur, pos, qin, NTB * D_MODEL);                  // q = k = src + pos
        GEMM(qin, P[0], P[1], Qb, NTB, 256, 256, 0);
        GEMM(qin, P[2], P[3], Kb, NTB, 256, 256, 0);
        GEMM(cur, P[4], P[5], Vb, NTB, 256, 256, 0);       // v = src
        FLASH(Qb, Kb, Vb, att, T, T);
        GEMM(att, P[6], P[7], proj, NTB, 256, 256, 0);
        ADDLN(cur, proj, 1, P[8], P[9], x1, NTB);
        GEMM(x1, P[12], P[13], hid, NTB, 256, FF, 1);      // leaky relu
        GEMM(hid, P[14], P[15], proj, NTB, FF, 256, 0);
        ADDLN(x1, proj, 1, P[10], P[11], outb, NTB);
        cur = outb;
    }
    const float* memory = cur;
    ADD(memory, pos, mpos, NTB * D_MODEL);                 // key-with-pos for cross attn

    // ---------------- decoder ----------------
    float* edge = (float*)d_out + (size_t)Nq * BATCH * D_MODEL;
    const float* tc = tgt;
    for (int j = 0; j < ND; ++j) {
        const float* const* P = (const float* const*)(d_in + DEC0 + j * 26);
        // 0..15 as encoder; 16:caWq 17:cabq 18:caWk 19:cabk 20:caWv 21:cabv
        // 22:caWo 23:cabo 24:ln3s 25:ln3b
        float* outb = (j & 1) ? tB : tA;
        // self attention
        ADD(tc, query_pos, qind, NQB * D_MODEL);
        GEMM(qind, P[0], P[1], Qd, NQB, 256, 256, 0);
        GEMM(qind, P[2], P[3], Kd, NQB, 256, 256, 0);
        GEMM(tc,   P[4], P[5], Vd, NQB, 256, 256, 0);
        FLASH(Qd, Kd, Vd, attd, Nq, Nq);
        if (j == ND - 1)
            edge_kernel<<<dim3(Nq, BATCH), 128, 0, stream>>>(Qd, Kd, edge, Nq);
        GEMM(attd, P[6], P[7], prjd, NQB, 256, 256, 0);
        ADDLN(tc, prjd, 1, P[8], P[9], x1d, NQB);
        // cross attention
        ADD(x1d, query_pos, qind, NQB * D_MODEL);
        GEMM(qind,   P[16], P[17], Qd, NQB, 256, 256, 0);
        GEMM(mpos,   P[18], P[19], Kb, NTB, 256, 256, 0);  // key = memory + pos
        GEMM(memory, P[20], P[21], Vb, NTB, 256, 256, 0);  // value = memory
        FLASH(Qd, Kb, Vb, attd, Nq, T);
        GEMM(attd, P[22], P[23], prjd, NQB, 256, 256, 0);
        ADDLN(x1d, prjd, 1, P[10], P[11], x2d, NQB);
        // FFN
        GEMM(x2d,  P[12], P[13], hidd, NQB, 256, FF, 1);
        GEMM(hidd, P[14], P[15], prjd, NQB, FF, 256, 0);
        ADDLN(x2d, prjd, 1, P[24], P[25], outb, NQB);
        tc = outb;
    }

    // final LayerNorm -> hs [1,Nq,B,D]
    ADDLN(tc, nullptr, 0, dn_s, dn_b, (float*)d_out, NQB);
}
